// MusicAttentionBlock_36816459661838
// MI455X (gfx1250) — matrix operand order repaired and checked by reading, not compiled, not run on hardware
//
#include <hip/hip_runtime.h>

// ---------------------------------------------------------------- constants
#define BB   2
#define SS   2048
#define DD   1024
#define HH   16
#define DHH  64
#define BSS  (BB * SS)           // 4096 rows
#define MAXD 128

typedef __attribute__((ext_vector_type(16))) __bf16 v16bf;
typedef __attribute__((ext_vector_type(8)))  float  v8f;

union BF16x16 { v16bf v; uint4 u[2]; unsigned short s[16]; };
union BF8     { uint4 u; unsigned short s[8]; };

__device__ __forceinline__ unsigned short f2bfbits(float f) {
    union { float f; unsigned u; } a; a.f = f;
    unsigned r = a.u + 0x7FFFu + ((a.u >> 16) & 1u);
    return (unsigned short)(r >> 16);
}
__device__ __forceinline__ __bf16 f2bf(float f) {
    union { unsigned short s; __bf16 b; } o; o.s = f2bfbits(f);
    return o.b;
}
__device__ __forceinline__ v8f wmma_bf16(v16bf a, v16bf b, v8f c) {
    return __builtin_amdgcn_wmma_f32_16x16x32_bf16(false, a, false, b,
                                                   (short)0, c, false, false);
}
__device__ __forceinline__ float redmax16(float v) {
    v = fmaxf(v, __shfl_xor(v, 1));
    v = fmaxf(v, __shfl_xor(v, 2));
    v = fmaxf(v, __shfl_xor(v, 4));
    v = fmaxf(v, __shfl_xor(v, 8));
    return v;
}
__device__ __forceinline__ float redsum16(float v) {
    v += __shfl_xor(v, 1);
    v += __shfl_xor(v, 2);
    v += __shfl_xor(v, 4);
    v += __shfl_xor(v, 8);
    return v;
}
// issue 64B (4 x b128) global -> LDS async copy; INST_OFFSET advances both sides
__device__ __forceinline__ void async_copy64(unsigned lds_off, const __bf16* gsrc) {
    asm volatile(
        "global_load_async_to_lds_b128 %0, %1, off\n\t"
        "global_load_async_to_lds_b128 %0, %1, off offset:16\n\t"
        "global_load_async_to_lds_b128 %0, %1, off offset:32\n\t"
        "global_load_async_to_lds_b128 %0, %1, off offset:48"
        :: "v"(lds_off), "v"(gsrc) : "memory");
}

// ---------------------------------------------------------------- LayerNorm -> bf16
__global__ void ln_bf16_kernel(const float* __restrict__ x,
                               const float* __restrict__ g,
                               const float* __restrict__ bt,
                               __bf16* __restrict__ out) {
    __shared__ float s1[256], s2[256];
    int row = blockIdx.x, t = threadIdx.x;
    const float* xr = x + (size_t)row * DD;
    int i0 = t * 4;
    float4 v = *(const float4*)(xr + i0);
    float va[4] = {v.x, v.y, v.z, v.w};
    float a = va[0] + va[1] + va[2] + va[3];
    float b = va[0]*va[0] + va[1]*va[1] + va[2]*va[2] + va[3]*va[3];
    s1[t] = a; s2[t] = b;
    __syncthreads();
    for (int o = 128; o > 0; o >>= 1) {
        if (t < o) { s1[t] += s1[t + o]; s2[t] += s2[t + o]; }
        __syncthreads();
    }
    float mu = s1[0] * (1.0f / DD);
    float var = s2[0] * (1.0f / DD) - mu * mu;
    float rstd = rsqrtf(var + 1e-5f);
    #pragma unroll
    for (int j = 0; j < 4; ++j)
        out[(size_t)row * DD + i0 + j] =
            f2bf((va[j] - mu) * rstd * g[i0 + j] + bt[i0 + j]);
}

// ---------------------------------------------------------------- f32 -> bf16 convert
__global__ void cvt_bf16_kernel(const float* __restrict__ in,
                                __bf16* __restrict__ out, int n) {
    int i = (blockIdx.x * 256 + threadIdx.x) * 4;
    if (i < n) {
        float4 v = *(const float4*)(in + i);
        out[i + 0] = f2bf(v.x); out[i + 1] = f2bf(v.y);
        out[i + 2] = f2bf(v.z); out[i + 3] = f2bf(v.w);
    }
}

// ---------------------------------------------------------------- WMMA GEMM
// C[M,N] = A[M,K](bf16,row) * B[K,N](bf16,row)
// block tile 128x64, BK=64, double-buffered LDS:
//   A staged global->LDS via async-to-LDS (ASYNCcnt), B staged transposed
//   through registers software-pipelined around the wmma block.
// EPI 0: +bias +resid -> f32 outf     (out-proj, ffn2)
// EPI 1: +bias, GELU -> bf16 outb     (ffn1)
// EPI 2: qkv scatter -> bf16 q[B,H,S,DH], k[B,H,S,DH], vT[B,H,DH,S]
template <int EPI>
__global__ void gemm_wmma(const __bf16* __restrict__ A, const __bf16* __restrict__ Bm,
                          const float* __restrict__ bias, const float* __restrict__ resid,
                          float* __restrict__ outf, __bf16* __restrict__ outb,
                          __bf16* __restrict__ qb, __bf16* __restrict__ kb,
                          __bf16* __restrict__ vtb, int M, int N, int K) {
    __shared__ __align__(16) unsigned short lds_a[2][128 * 72];  // [row][k], pad 72
    __shared__ __align__(16) unsigned short lds_b[2][64 * 72];   // transposed [n][k]

    const int wave = threadIdx.x >> 5, lane = threadIdx.x & 31;
    const int half = lane >> 4, l16 = lane & 15;
    const int wm = wave >> 1, wn = wave & 1;
    const int bm = blockIdx.y * 128, bn = blockIdx.x * 64;

    const int ar = threadIdx.x >> 1, ahf = threadIdx.x & 1;   // A: 64B per thread
    const int bkr = threadIdx.x >> 2, bseg = threadIdx.x & 3; // B: 16 elems per thread

    v8f acc[2][2] = {};

    // ---- prologue: stage k-tile 0 into buffer 0
    {
        async_copy64((unsigned)(size_t)&lds_a[0][ar * 72 + ahf * 32],
                     A + (size_t)(bm + ar) * K + ahf * 32);
        const uint4* gB = (const uint4*)(Bm + (size_t)bkr * N + bn + bseg * 16);
        BF8 u0, u1; u0.u = gB[0]; u1.u = gB[1];
        #pragma unroll
        for (int j = 0; j < 8; ++j) {
            lds_b[0][(bseg * 16 + j) * 72 + bkr]     = u0.s[j];
            lds_b[0][(bseg * 16 + 8 + j) * 72 + bkr] = u1.s[j];
        }
    }
    asm volatile("s_wait_asynccnt 0" ::: "memory");
    __syncthreads();

    const int NT = K >> 6;
    for (int t = 0; t < NT; ++t) {
        const int cur = t & 1, nxt = cur ^ 1;
        const bool more = (t + 1 < NT);
        uint4 br0, br1;
        if (more) {   // overlap: async A(t+1) + B(t+1) into regs while computing t
            int kt = (t + 1) << 6;
            async_copy64((unsigned)(size_t)&lds_a[nxt][ar * 72 + ahf * 32],
                         A + (size_t)(bm + ar) * K + kt + ahf * 32);
            const uint4* gB = (const uint4*)(Bm + (size_t)(kt + bkr) * N + bn + bseg * 16);
            br0 = gB[0]; br1 = gB[1];
        }
        // compute on cur: two K=32 sub-steps, 8 wmma per barrier
        #pragma unroll
        for (int ks = 0; ks < 64; ks += 32) {
            BF16x16 af[2], bfm[2];
            #pragma unroll
            for (int mt = 0; mt < 2; ++mt) {   // A frag: lane row m, K=16*half+e
                int m = wm * 32 + mt * 16 + l16;
                af[mt].u[0] = *(const uint4*)&lds_a[cur][m * 72 + ks + half * 16];
                af[mt].u[1] = *(const uint4*)&lds_a[cur][m * 72 + ks + half * 16 + 8];
            }
            #pragma unroll
            for (int nn = 0; nn < 2; ++nn) {   // B frag: lane col n, K=16*half+e
                int n = wn * 32 + nn * 16 + l16;
                const uint4* p = (const uint4*)&lds_b[cur][n * 72 + ks + half * 16];
                bfm[nn].u[0] = p[0]; bfm[nn].u[1] = p[1];
            }
            #pragma unroll
            for (int mt = 0; mt < 2; ++mt)
                #pragma unroll
                for (int nn = 0; nn < 2; ++nn)
                    acc[mt][nn] = wmma_bf16(af[mt].v, bfm[nn].v, acc[mt][nn]);
        }
        if (more) {   // transpose-store B regs into the other buffer
            BF8 u0, u1; u0.u = br0; u1.u = br1;
            #pragma unroll
            for (int j = 0; j < 8; ++j) {
                lds_b[nxt][(bseg * 16 + j) * 72 + bkr]     = u0.s[j];
                lds_b[nxt][(bseg * 16 + 8 + j) * 72 + bkr] = u1.s[j];
            }
        }
        asm volatile("s_wait_asynccnt 0" ::: "memory");
        __syncthreads();
    }

    const int mbase = bm + wm * 32, nbase = bn + wn * 32;
    #pragma unroll
    for (int mt = 0; mt < 2; ++mt)
        #pragma unroll
        for (int nn = 0; nn < 2; ++nn)
            #pragma unroll
            for (int r = 0; r < 8; ++r) {
                int m = mbase + mt * 16 + r + 8 * half;
                int n = nbase + nn * 16 + l16;
                float v = acc[mt][nn][r];
                if constexpr (EPI == 0) {
                    v += bias[n] + resid[(size_t)m * N + n];
                    outf[(size_t)m * N + n] = v;
                } else if constexpr (EPI == 1) {
                    v += bias[n];
                    v = 0.5f * v * (1.0f + erff(v * 0.70710678118f));
                    outb[(size_t)m * N + n] = f2bf(v);
                } else {
                    int which = n >> 10, rr = n & 1023;
                    int hh = rr >> 6, dh = rr & 63;
                    int b = m >> 11, s = m & 2047;
                    size_t bhs = ((size_t)b * HH + hh);
                    if (which == 0)
                        qb[(bhs * SS + s) * DHH + dh] = f2bf(v);
                    else if (which == 1)
                        kb[(bhs * SS + s) * DHH + dh] = f2bf(v);
                    else
                        vtb[(bhs * DHH + dh) * SS + s] = f2bf(v);
                }
            }
}

// ---------------------------------------------------------------- flash attention
// grid (S/64, B*H), 128 threads = 4 independent waves, each a 16-row q tile.
__global__ void attn_kernel(const __bf16* __restrict__ qb, const __bf16* __restrict__ kb,
                            const __bf16* __restrict__ vtb, const float* __restrict__ rel,
                            __bf16* __restrict__ attn_out) {
    __shared__ __align__(16) unsigned short pl[4][16 * 32];  // per-wave P staging

    const int wave = threadIdx.x >> 5, lane = threadIdx.x & 31;
    const int half = lane >> 4, l16 = lane & 15;
    const int bh = blockIdx.y;              // b*H + h
    const int h = bh & (HH - 1), b = bh >> 4;
    const int q0 = blockIdx.x * 64 + wave * 16;
    const int qmax = q0 + 15;
    const float scale = 0.125f;             // 1/sqrt(64)

    // Q fragments (A layout), resident for whole loop
    const __bf16* qrow = qb + ((size_t)bh * SS + q0 + l16) * DHH;
    BF16x16 qa[2];
    #pragma unroll
    for (int f = 0; f < 2; ++f) {
        int base = f * 32 + half * 16;
        qa[f].u[0] = *(const uint4*)(qrow + base);
        qa[f].u[1] = *(const uint4*)(qrow + base + 8);
    }

    float mst[8], lst[8];
    #pragma unroll
    for (int r = 0; r < 8; ++r) { mst[r] = -1e30f; lst[r] = 0.0f; }
    v8f o[4] = {};

    unsigned short* pw = &pl[wave][0];

    for (int c0 = 0; c0 <= qmax; c0 += 32) {
        v8f st[2];
        #pragma unroll
        for (int t = 0; t < 2; ++t) {
            int kv0 = c0 + t * 16;
            v8f z = {};
            if (kv0 <= qmax) {   // S = Q . K^T : B frag lane n = kv pos, contiguous dh
                const __bf16* kr = kb + ((size_t)bh * SS + kv0 + l16) * DHH + 16 * half;
                BF16x16 kf;
                kf.u[0] = *(const uint4*)(kr);
                kf.u[1] = *(const uint4*)(kr + 8);
                z = wmma_bf16(qa[0].v, kf.v, z);
                kf.u[0] = *(const uint4*)(kr + 32);
                kf.u[1] = *(const uint4*)(kr + 40);
                z = wmma_bf16(qa[1].v, kf.v, z);
            }
            #pragma unroll
            for (int r = 0; r < 8; ++r) {   // scale + rel bias + causal mask
                int i = q0 + r + 8 * half;
                int j = kv0 + l16;
                float sv;
                if (j <= i) {
                    int d = j - i; if (d < -MAXD) d = -MAXD;
                    sv = z[r] * scale + rel[(d + MAXD) * HH + h];
                } else sv = -1e30f;
                st[t][r] = sv;
            }
        }
        // online softmax over the 32-key chunk
        #pragma unroll
        for (int r = 0; r < 8; ++r) {
            float rm = redmax16(fmaxf(st[0][r], st[1][r]));
            float nm = fmaxf(mst[r], rm);
            float alpha = __expf(mst[r] - nm);
            float p0 = __expf(st[0][r] - nm);
            float p1 = __expf(st[1][r] - nm);
            float rs = redsum16(p0 + p1);
            lst[r] = lst[r] * alpha + rs;
            mst[r] = nm;
            #pragma unroll
            for (int nt = 0; nt < 4; ++nt) o[nt][r] *= alpha;
            // P: C layout (row r+8*half, col l16) -> LDS as bf16
            pw[(r + 8 * half) * 32 + l16]      = f2bfbits(p0);
            pw[(r + 8 * half) * 32 + 16 + l16] = f2bfbits(p1);
        }
        asm volatile("s_wait_dscnt 0" ::: "memory");  // intra-wave LDS RAW fence
        // reload P in A layout: lane row l16, K = kv within chunk
        BF16x16 pf;
        pf.u[0] = *(const uint4*)&pw[l16 * 32 + half * 16];
        pf.u[1] = *(const uint4*)&pw[l16 * 32 + half * 16 + 8];
        // O += P . V  (V^T layout gives contiguous B frags: lane n = dh)
        #pragma unroll
        for (int nt = 0; nt < 4; ++nt) {
            const __bf16* vr = vtb + ((size_t)bh * DHH + nt * 16 + l16) * SS
                                   + c0 + 16 * half;
            BF16x16 vf;
            vf.u[0] = *(const uint4*)(vr);
            vf.u[1] = *(const uint4*)(vr + 8);
            o[nt] = wmma_bf16(pf.v, vf.v, o[nt]);
        }
    }

    // finalize: attn_out[b, i, h*64+dh] = O / l   (bf16 for the out-proj GEMM)
    #pragma unroll
    for (int nt = 0; nt < 4; ++nt)
        #pragma unroll
        for (int r = 0; r < 8; ++r) {
            int i = q0 + r + 8 * half;
            int col = h * DHH + nt * 16 + l16;
            attn_out[((size_t)b * SS + i) * DD + col] = f2bf(o[nt][r] / lst[r]);
        }
}

// ---------------------------------------------------------------- launch
extern "C" void kernel_launch(void* const* d_in, const int* in_sizes, int n_in,
                              void* d_out, int out_size, void* d_ws, size_t ws_size,
                              hipStream_t stream) {
    (void)in_sizes; (void)n_in; (void)out_size; (void)ws_size;
    const float* x     = (const float*)d_in[0];
    const float* ln1_g = (const float*)d_in[1];
    const float* ln1_b = (const float*)d_in[2];
    const float* w_qkv = (const float*)d_in[3];
    const float* w_out = (const float*)d_in[4];
    const float* b_out = (const float*)d_in[5];
    const float* rel   = (const float*)d_in[6];
    const float* ln2_g = (const float*)d_in[7];
    const float* ln2_b = (const float*)d_in[8];
    const float* w1    = (const float*)d_in[9];
    const float* b1    = (const float*)d_in[10];
    const float* w2    = (const float*)d_in[11];
    const float* b2    = (const float*)d_in[12];
    float* outp = (float*)d_out;

    char* p = (char*)d_ws;
    auto carve = [&](size_t bytes) { char* r = p; p += (bytes + 255) & ~size_t(255); return r; };
    __bf16* xn_bf   = (__bf16*)carve((size_t)BSS * DD * 2);
    __bf16* wqkv_bf = (__bf16*)carve((size_t)DD * 3 * DD * 2);
    __bf16* wout_bf = (__bf16*)carve((size_t)DD * DD * 2);
    __bf16* w1_bf   = (__bf16*)carve((size_t)DD * 2 * DD * 2);
    __bf16* w2_bf   = (__bf16*)carve((size_t)2 * DD * DD * 2);
    __bf16* q_bf    = (__bf16*)carve((size_t)BSS * DD * 2);
    __bf16* k_bf    = (__bf16*)carve((size_t)BSS * DD * 2);
    __bf16* vt_bf   = (__bf16*)carve((size_t)BSS * DD * 2);
    __bf16* attn_bf = (__bf16*)carve((size_t)BSS * DD * 2);
    float*  x2_f    = (float*) carve((size_t)BSS * DD * 4);
    __bf16* h2_bf   = (__bf16*)carve((size_t)BSS * DD * 2);
    __bf16* ffn1_bf = (__bf16*)carve((size_t)BSS * 2 * DD * 2);

    // 1. LN1 -> bf16
    ln_bf16_kernel<<<BSS, 256, 0, stream>>>(x, ln1_g, ln1_b, xn_bf);
    // 2. weight converts
    cvt_bf16_kernel<<<(DD * 3 * DD / 4 + 255) / 256, 256, 0, stream>>>(w_qkv, wqkv_bf, DD * 3 * DD);
    cvt_bf16_kernel<<<(DD * DD / 4 + 255) / 256, 256, 0, stream>>>(w_out, wout_bf, DD * DD);
    cvt_bf16_kernel<<<(DD * 2 * DD / 4 + 255) / 256, 256, 0, stream>>>(w1, w1_bf, DD * 2 * DD);
    cvt_bf16_kernel<<<(2 * DD * DD / 4 + 255) / 256, 256, 0, stream>>>(w2, w2_bf, 2 * DD * DD);
    // 3. QKV GEMM (scatter to head layouts, V transposed)
    gemm_wmma<2><<<dim3(3 * DD / 64, BSS / 128), 256, 0, stream>>>(
        xn_bf, wqkv_bf, nullptr, nullptr, nullptr, nullptr,
        q_bf, k_bf, vt_bf, BSS, 3 * DD, DD);
    // 4. flash attention
    attn_kernel<<<dim3(SS / 64, BB * HH), 128, 0, stream>>>(q_bf, k_bf, vt_bf, rel, attn_bf);
    // 5. out-proj + bias + residual(x) -> x2 (f32)
    gemm_wmma<0><<<dim3(DD / 64, BSS / 128), 256, 0, stream>>>(
        attn_bf, wout_bf, b_out, x, x2_f, nullptr, nullptr, nullptr, nullptr,
        BSS, DD, DD);
    // 6. LN2 -> bf16
    ln_bf16_kernel<<<BSS, 256, 0, stream>>>(x2_f, ln2_g, ln2_b, h2_bf);
    // 7. FFN1 + bias + GELU -> bf16
    gemm_wmma<1><<<dim3(2 * DD / 64, BSS / 128), 256, 0, stream>>>(
        h2_bf, w1_bf, b1, nullptr, nullptr, ffn1_bf, nullptr, nullptr, nullptr,
        BSS, 2 * DD, DD);
    // 8. FFN2 + bias + residual(x2) -> d_out (f32)
    gemm_wmma<0><<<dim3(DD / 64, BSS / 128), 256, 0, stream>>>(
        ffn1_bf, w2_bf, b2, x2_f, outp, nullptr, nullptr, nullptr, nullptr,
        BSS, DD, 2 * DD);
}
